// DynamicTree2_1829656068777
// MI455X (gfx1250) — compile-verified, matrix operand order checked by reading, not run on hardware
//
#include <hip/hip_runtime.h>
#include <hip/hip_bf16.h>
#include <string.h>

#define NPTS 131072
#define BATCH 2
#define FEATC 69
#define KERN 128
#define MOUT 32768
#define CH 432            // activation channel stride in halves (16B-aligned rows)
#define MAX_KT 40

typedef __attribute__((ext_vector_type(16))) _Float16 v16h;
typedef __attribute__((ext_vector_type(8)))  _Float16 v8h;
typedef __attribute__((ext_vector_type(8)))  float    v8f;

// ---------------- sort: stable argsort via bitonic on (key<<32|idx) ----------------
__global__ void sort_init(const float* __restrict__ inputs, unsigned long long* __restrict__ keys)
{
    int n  = blockIdx.x * blockDim.x + threadIdx.x;
    int bc = blockIdx.y;                 // b*2 + ch
    int b = bc >> 1, ch = bc & 1;
    float v = inputs[((size_t)b * NPTS + n) * FEATC + 3 + ch];
    unsigned int bits = __float_as_uint(v);
    unsigned int k = (bits & 0x80000000u) ? ~bits : (bits | 0x80000000u);
    keys[(size_t)bc * NPTS + n] = ((unsigned long long)k << 32) | (unsigned int)n;
}

__global__ void bitonic_pass(unsigned long long* __restrict__ keys, int k, int j)
{
    int i  = blockIdx.x * blockDim.x + threadIdx.x;
    unsigned long long* a = keys + (size_t)blockIdx.y * NPTS;
    int ixj = i ^ j;
    if (ixj > i) {
        unsigned long long x = a[i], y = a[ixj];
        bool up = ((i & k) == 0);
        if ((x > y) == up) { a[i] = y; a[ixj] = x; }
    }
}

// ---------------- gather sorted rows into f16 activation buffer ----------------
__global__ void gather_kernel(const float* __restrict__ inputs,
                              const unsigned long long* __restrict__ keys,
                              _Float16* __restrict__ act)
{
    int n = blockIdx.x * blockDim.x + threadIdx.x;
    int b = blockIdx.y;
    unsigned int idx1 = (unsigned int)keys[(size_t)(b * 2 + 1) * NPTS + n];
    unsigned int idx0 = (unsigned int)keys[(size_t)(b * 2 + 0) * NPTS + n];
    const float* r1 = inputs + ((size_t)b * NPTS + idx1) * FEATC;
    const float* r0 = inputs + ((size_t)b * NPTS + idx0) * FEATC;
    _Float16* o = act + ((size_t)b * NPTS + n) * CH;
    for (int c = 0; c < FEATC; ++c) o[16  + c] = (_Float16)r1[c];   // x0
    for (int c = 0; c < FEATC; ++c) o[224 + c] = (_Float16)r0[c];   // g0
    for (int c = 4;   c < 16;  ++c) o[c] = (_Float16)0.f;           // pads
    for (int c = 85;  c < 96;  ++c) o[c] = (_Float16)0.f;
    for (int c = 293; c < 304; ++c) o[c] = (_Float16)0.f;
}

// ---------------- weight -> WMMA B-fragment swizzle (zero-padded) ----------------
struct SwzSpec {           // up to 2 valid ranges per 32-wide K tile
    short lo0[MAX_KT], hi0[MAX_KT], w0[MAX_KT];
    short lo1[MAX_KT], hi1[MAX_KT], w1[MAX_KT];
    short dsel[MAX_KT];
    int kt, nOut, cinTot;  // weight layout w[(d*cinTot + wc)*nOut + n]
};

__global__ void swizzleB(const float* __restrict__ w, _Float16* __restrict__ dst, SwzSpec sp)
{
    int tid = blockIdx.x * blockDim.x + threadIdx.x;
    int nt = sp.nOut >> 4;
    int total = sp.kt * nt * 32;
    if (tid >= total) return;
    int lane = tid & 31;
    int rest = tid >> 5;
    int ntile = rest % nt;
    int t = rest / nt;
    int half_ = lane >> 4, col = lane & 15;
    int n = (ntile << 4) + col;
    union { _Float16 h[16]; v16h v; } u;
    for (int i = 0; i < 16; ++i) {
        int koff = (i < 8 ? i : i + 8) + (half_ << 3);   // K position in 32-wide tile
        int wc = -1;
        if (koff >= sp.lo0[t] && koff < sp.hi0[t]) wc = sp.w0[t] + (koff - sp.lo0[t]);
        else if (koff >= sp.lo1[t] && koff < sp.hi1[t]) wc = sp.w1[t] + (koff - sp.lo1[t]);
        float v = 0.f;
        if (wc >= 0) v = w[((size_t)sp.dsel[t] * sp.cinTot + wc) * sp.nOut + n];
        u.h[i] = (_Float16)v;
    }
    ((v16h*)dst)[(size_t)(t * nt + ntile) * 32 + lane] = u.v;
}

// ---------------- per-layer compile-time shapes ----------------
struct LPre0 {   // conv k3 SAME over x0(69) -> convA
    static constexpr int KT = 9, STRIDEM = 1, DOFF = -1, DMAX = 2;
    static constexpr int LEN_IN = NPTS, LEN_OUT = NPTS;
    static constexpr int IN_STRIDE = CH, OUT_STRIDE = CH, OUT_BASE = 304;
    static constexpr bool NEVER_OOB = false;
    static constexpr int CB[9] = {16,48,80, 16,48,80, 16,48,80};
    static constexpr int DD[9] = { 0, 0, 0,  1, 1, 1,  2, 2, 2};
};
struct LPre1 {   // conv k3 SAME over [x0(69)|convA(128)] -> convB
    static constexpr int KT = 21, STRIDEM = 1, DOFF = -1, DMAX = 2;
    static constexpr int LEN_IN = NPTS, LEN_OUT = NPTS;
    static constexpr int IN_STRIDE = CH, OUT_STRIDE = CH, OUT_BASE = 96;
    static constexpr bool NEVER_OOB = false;
    static constexpr int CB[21] = {16,48,80,304,336,368,400,
                                   16,48,80,304,336,368,400,
                                   16,48,80,304,336,368,400};
    static constexpr int DD[21] = {0,0,0,0,0,0,0, 1,1,1,1,1,1,1, 2,2,2,2,2,2,2};
};
struct LMerge {  // conv k4 stride4 VALID over [F|x0|convB|g0] -> S
    static constexpr int KT = 40, STRIDEM = 4, DOFF = 0, DMAX = 3;
    static constexpr int LEN_IN = NPTS, LEN_OUT = MOUT;
    static constexpr int IN_STRIDE = CH, OUT_STRIDE = KERN, OUT_BASE = 0;
    static constexpr bool NEVER_OOB = true;
    static constexpr int CB[40] = {0,32,64,96,128,160,192,224,256,288,
                                   0,32,64,96,128,160,192,224,256,288,
                                   0,32,64,96,128,160,192,224,256,288,
                                   0,32,64,96,128,160,192,224,256,288};
    static constexpr int DD[40] = {0,0,0,0,0,0,0,0,0,0, 1,1,1,1,1,1,1,1,1,1,
                                   2,2,2,2,2,2,2,2,2,2, 3,3,3,3,3,3,3,3,3,3};
};
struct LPost {   // conv k3 SAME over 128 -> 128
    static constexpr int KT = 12, STRIDEM = 1, DOFF = -1, DMAX = 2;
    static constexpr int LEN_IN = MOUT, LEN_OUT = MOUT;
    static constexpr int IN_STRIDE = KERN, OUT_STRIDE = KERN, OUT_BASE = 0;
    static constexpr bool NEVER_OOB = false;
    static constexpr int CB[12] = {0,32,64,96, 0,32,64,96, 0,32,64,96};
    static constexpr int DD[12] = {0,0,0,0, 1,1,1,1, 2,2,2,2};
};

// ---------------- device helpers ----------------
static __device__ __forceinline__ v16h loadA16(const _Float16* __restrict__ p)
{
    v8h lo = *(const v8h*)(p);
    v8h hi = *(const v8h*)(p + 16);
    return __builtin_shufflevector(lo, hi, 0,1,2,3,4,5,6,7,8,9,10,11,12,13,14,15);
}

static __device__ __forceinline__ v16h zero16()
{
    union { unsigned long long u[4]; v16h v; } z;
    z.u[0] = 0; z.u[1] = 0; z.u[2] = 0; z.u[3] = 0;
    return z.v;
}

template<class L, bool CHECK>
static __device__ __forceinline__ void gemm_accum(
    const _Float16* __restrict__ actB, const _Float16* __restrict__ bw,
    int m0, int mrow, int half_, int lane, v8f acc[8])
{
    #pragma unroll
    for (int t = 0; t < L::KT; ++t) {
        const int d = L::DD[t], cb = L::CB[t];
        int src = (m0 + mrow) * L::STRIDEM + d + L::DOFF;
        bool inb = true;
        int srcC = src;
        if (CHECK) {
            inb = (src >= 0) && (src < L::LEN_IN);
            srcC = inb ? src : 0;
        }
        const _Float16* p = actB + (size_t)srcC * L::IN_STRIDE + cb + (half_ << 3);
        v16h a = loadA16(p);
        if (CHECK) { if (!inb) a = zero16(); }
        const v16h* bp = (const v16h*)bw + ((size_t)t << 8) + lane;  // (t*8+n)*32 + lane
        #pragma unroll
        for (int n = 0; n < 8; ++n) {
            acc[n] = __builtin_amdgcn_wmma_f32_16x16x32_f16(
                false, a, false, bp[n << 5], (short)0, acc[n], false, false);
        }
    }
}

// ---------------- generic conv-as-GEMM: one wave per 16-row tile, 128 outs ----------------
template<class L>
__global__ __launch_bounds__(256) void gemm_conv(
    const _Float16* __restrict__ act, const _Float16* __restrict__ bw,
    const float* __restrict__ bias, _Float16* __restrict__ outAct)
{
    int wave = (int)((blockIdx.x * blockDim.x + threadIdx.x) >> 5);
    int lane = (int)(threadIdx.x & 31);
    int b = blockIdx.y;
    int tilesM = L::LEN_OUT >> 4;
    if (wave >= tilesM) return;                 // uniform per wave
    int m0 = wave << 4;
    const _Float16* actB = act + (size_t)b * L::LEN_IN * L::IN_STRIDE;
    _Float16* outB = outAct + (size_t)b * L::LEN_OUT * L::OUT_STRIDE;
    int half_ = lane >> 4, mrow = lane & 15, col0 = lane & 15;

    v8f acc[8];
    #pragma unroll
    for (int n = 0; n < 8; ++n)
        #pragma unroll
        for (int i = 0; i < 8; ++i) acc[n][i] = 0.0f;

    if constexpr (L::NEVER_OOB) {
        gemm_accum<L, false>(actB, bw, m0, mrow, half_, lane, acc);
    } else {
        int minSrc = m0 * L::STRIDEM + L::DOFF;
        int maxSrc = (m0 + 15) * L::STRIDEM + L::DMAX + L::DOFF;
        if (minSrc >= 0 && maxSrc < L::LEN_IN)
            gemm_accum<L, false>(actB, bw, m0, mrow, half_, lane, acc);  // interior: no checks
        else
            gemm_accum<L, true>(actB, bw, m0, mrow, half_, lane, acc);   // edge waves only
    }

    #pragma unroll
    for (int n = 0; n < 8; ++n) {
        float bv = bias[(n << 4) + col0];
        #pragma unroll
        for (int r = 0; r < 8; ++r) {
            float v = acc[n][r] + bv;
            v = v > 0.f ? v : 0.f;              // relu
            int row = m0 + r + (half_ << 3);
            outB[(size_t)row * L::OUT_STRIDE + L::OUT_BASE + (n << 4) + col0] = (_Float16)v;
        }
    }
}

// ---------------- flags: X4 @ w_flags (N=4) + sigmoid, writes F and Fd ----------------
__global__ void flags_kernel(_Float16* __restrict__ act, const float* __restrict__ wf,
                             const float* __restrict__ bf, float* __restrict__ out)
{
    int n = blockIdx.x * blockDim.x + threadIdx.x;
    int b = blockIdx.y;
    if (n >= NPTS) return;
    _Float16* row = act + ((size_t)b * NPTS + n) * CH;
    float s0 = bf[0], s1 = bf[1], s2 = bf[2], s3 = bf[3];
    int wc = 0;
    auto accum = [&](int c0, int c1) {
        for (int c = c0; c < c1; ++c, ++wc) {
            float x = (float)row[c];
            const float* wr = wf + wc * 4;
            s0 += x * wr[0]; s1 += x * wr[1]; s2 += x * wr[2]; s3 += x * wr[3];
        }
    };
    accum(16, 85);    // x0   -> wc 0..68
    accum(96, 224);   // convB-> wc 69..196
    accum(224, 293);  // g0   -> wc 197..265
    float F0 = 1.f / (1.f + __expf(-s0));
    float F1 = 1.f / (1.f + __expf(-s1));
    float F2 = 1.f / (1.f + __expf(-s2));
    float F3 = 1.f / (1.f + __expf(-s3));
    row[0] = (_Float16)F0; row[1] = (_Float16)F1;
    row[2] = (_Float16)F2; row[3] = (_Float16)F3;
    int m = n >> 2, j = n & 3;
    out[((size_t)b * MOUT + m) * 20 + j] = F0 - F1;
}

// ---------------- sym: S @ w_sym (16 bins) via WMMA + softmax epilogue ----------------
__global__ __launch_bounds__(256) void sym_softmax(
    const _Float16* __restrict__ S, const _Float16* __restrict__ bw,
    const float* __restrict__ bias, float* __restrict__ out)
{
    int wave = (int)((blockIdx.x * blockDim.x + threadIdx.x) >> 5);
    int lane = (int)(threadIdx.x & 31);
    int b = blockIdx.y;
    int tilesM = MOUT >> 4;
    if (wave >= tilesM) return;
    int m0 = wave << 4;
    const _Float16* Sb = S + (size_t)b * MOUT * KERN;
    int half_ = lane >> 4, mrow = lane & 15, col = lane & 15;

    v8f acc;
    #pragma unroll
    for (int i = 0; i < 8; ++i) acc[i] = 0.0f;
    #pragma unroll
    for (int t = 0; t < 4; ++t) {
        const _Float16* p = Sb + (size_t)(m0 + mrow) * KERN + t * 32 + (half_ << 3);
        v16h a = loadA16(p);
        v16h bf = ((const v16h*)bw)[(t << 5) + lane];
        acc = __builtin_amdgcn_wmma_f32_16x16x32_f16(false, a, false, bf, (short)0, acc, false, false);
    }
    float bv = bias[col];
    #pragma unroll
    for (int r = 0; r < 8; ++r) {
        float v = acc[r] + bv;
        float mx = v;
        #pragma unroll
        for (int o = 8; o >= 1; o >>= 1) mx = fmaxf(mx, __shfl_xor(mx, o, 16));
        float e = __expf(v - mx);
        float sm = e;
        #pragma unroll
        for (int o = 8; o >= 1; o >>= 1) sm += __shfl_xor(sm, o, 16);
        int row = m0 + r + (half_ << 3);
        out[((size_t)b * MOUT + row) * 20 + 4 + col] = e / sm;
    }
}

// ---------------- host side ----------------
struct Range { int lo, hi, w; };   // act channels [lo,hi) -> weight channel w + (c-lo)

static void fillSwzTile(SwzSpec& s, int t, int cb, const Range* rg, int nr)
{
    int found = 0;
    for (int i = 0; i < nr; ++i) {
        int lo = rg[i].lo > cb ? rg[i].lo : cb;
        int hi = rg[i].hi < cb + 32 ? rg[i].hi : cb + 32;
        if (lo < hi) {
            short l = (short)(lo - cb), h = (short)(hi - cb);
            short w = (short)(rg[i].w + (lo - rg[i].lo));
            if (found == 0) { s.lo0[t] = l; s.hi0[t] = h; s.w0[t] = w; }
            else if (found == 1) { s.lo1[t] = l; s.hi1[t] = h; s.w1[t] = w; }
            ++found;
        }
    }
}

extern "C" void kernel_launch(void* const* d_in, const int* in_sizes, int n_in,
                              void* d_out, int out_size, void* d_ws, size_t ws_size,
                              hipStream_t stream)
{
    (void)in_sizes; (void)n_in; (void)out_size; (void)ws_size;
    const float* inputs  = (const float*)d_in[0];
    const float* w_pre0  = (const float*)d_in[1];
    const float* b_pre0  = (const float*)d_in[2];
    const float* w_pre1  = (const float*)d_in[3];
    const float* b_pre1  = (const float*)d_in[4];
    const float* w_flags = (const float*)d_in[5];
    const float* b_flags = (const float*)d_in[6];
    const float* w_merge = (const float*)d_in[7];
    const float* b_merge = (const float*)d_in[8];
    const float* w_post[4] = { (const float*)d_in[9],  (const float*)d_in[11],
                               (const float*)d_in[13], (const float*)d_in[15] };
    const float* b_post[4] = { (const float*)d_in[10], (const float*)d_in[12],
                               (const float*)d_in[14], (const float*)d_in[16] };
    const float* w_sym = (const float*)d_in[17];
    const float* b_sym = (const float*)d_in[18];
    float* out = (float*)d_out;

    // workspace carve-up
    char* ws = (char*)d_ws;
    size_t off = 0;
    auto walloc = [&](size_t bytes) -> void* {
        void* p = ws + off;
        off = (off + bytes + 255) & ~(size_t)255;
        return p;
    };
    unsigned long long* keys = (unsigned long long*)walloc((size_t)4 * NPTS * 8);
    _Float16* ACT = (_Float16*)walloc((size_t)BATCH * NPTS * CH * 2);
    _Float16* S0  = (_Float16*)walloc((size_t)BATCH * MOUT * KERN * 2);
    _Float16* S1  = (_Float16*)walloc((size_t)BATCH * MOUT * KERN * 2);
    _Float16* BWpre0  = (_Float16*)walloc((size_t)9  * 8 * 32 * 32);
    _Float16* BWpre1  = (_Float16*)walloc((size_t)21 * 8 * 32 * 32);
    _Float16* BWmerge = (_Float16*)walloc((size_t)40 * 8 * 32 * 32);
    _Float16* BWpost[4];
    for (int i = 0; i < 4; ++i) BWpost[i] = (_Float16*)walloc((size_t)12 * 8 * 32 * 32);
    _Float16* BWsym = (_Float16*)walloc((size_t)4 * 1 * 32 * 32);

    // ---- build swizzle specs (tile order MUST match LPre0/LPre1/LMerge/LPost tables) ----
    SwzSpec sPre0, sPre1, sMerge, sPost, sSym;
    memset(&sPre0, 0, sizeof(sPre0)); memset(&sPre1, 0, sizeof(sPre1));
    memset(&sMerge, 0, sizeof(sMerge)); memset(&sPost, 0, sizeof(sPost));
    memset(&sSym, 0, sizeof(sSym));

    { // pre0: K = 3 taps x x0(69)
        Range rg[1] = { {16, 85, 0} };
        int t = 0;
        for (int d = 0; d < 3; ++d)
            for (int ct = 0; ct < 3; ++ct, ++t) {
                sPre0.dsel[t] = (short)d;
                fillSwzTile(sPre0, t, 16 + ct * 32, rg, 1);
            }
        sPre0.kt = 9; sPre0.nOut = 128; sPre0.cinTot = 69;
    }
    { // pre1: K = 3 taps x [x0(69) | convA(128)]
        Range rg[2] = { {16, 85, 0}, {304, 432, 69} };
        int bases[7] = { 16, 48, 80, 304, 336, 368, 400 };
        int t = 0;
        for (int d = 0; d < 3; ++d)
            for (int ct = 0; ct < 7; ++ct, ++t) {
                sPre1.dsel[t] = (short)d;
                fillSwzTile(sPre1, t, bases[ct], rg, 2);
            }
        sPre1.kt = 21; sPre1.nOut = 128; sPre1.cinTot = 197;
    }
    { // merge: K = 4 taps x [F(4) | x0(69) | convB(128) | g0(69)]
        Range rg[3] = { {0, 4, 0}, {16, 85, 4}, {96, 293, 73} };
        int t = 0;
        for (int d = 0; d < 4; ++d)
            for (int ct = 0; ct < 10; ++ct, ++t) {
                sMerge.dsel[t] = (short)d;
                fillSwzTile(sMerge, t, ct * 32, rg, 3);
            }
        sMerge.kt = 40; sMerge.nOut = 128; sMerge.cinTot = 270;
    }
    { // post: K = 3 taps x 128
        Range rg[1] = { {0, 128, 0} };
        int t = 0;
        for (int d = 0; d < 3; ++d)
            for (int ct = 0; ct < 4; ++ct, ++t) {
                sPost.dsel[t] = (short)d;
                fillSwzTile(sPost, t, ct * 32, rg, 1);
            }
        sPost.kt = 12; sPost.nOut = 128; sPost.cinTot = 128;
    }
    { // sym: 128 -> 16
        Range rg[1] = { {0, 128, 0} };
        for (int ct = 0; ct < 4; ++ct) {
            sSym.dsel[ct] = 0;
            fillSwzTile(sSym, ct, ct * 32, rg, 1);
        }
        sSym.kt = 4; sSym.nOut = 16; sSym.cinTot = 128;
    }

    dim3 blk(256);
    // 1. stable argsort of channels 3 & 4 (both batches): init + bitonic passes
    sort_init<<<dim3(NPTS / 256, 4), blk, 0, stream>>>(inputs, keys);
    for (int k = 2; k <= NPTS; k <<= 1)
        for (int j = k >> 1; j > 0; j >>= 1)
            bitonic_pass<<<dim3(NPTS / 256, 4), blk, 0, stream>>>(keys, k, j);

    // 2. gather sorted rows into activation buffer (f16)
    gather_kernel<<<dim3(NPTS / 256, BATCH), blk, 0, stream>>>(inputs, keys, ACT);

    // 3. weight swizzles
    auto swz = [&](const float* w, _Float16* dst, const SwzSpec& s) {
        int total = s.kt * (s.nOut >> 4) * 32;
        swizzleB<<<dim3((total + 255) / 256), blk, 0, stream>>>(w, dst, s);
    };
    swz(w_pre0, BWpre0, sPre0);
    swz(w_pre1, BWpre1, sPre1);
    swz(w_merge, BWmerge, sMerge);
    for (int i = 0; i < 4; ++i) swz(w_post[i], BWpost[i], sPost);
    swz(w_sym, BWsym, sSym);

    // 4. GEMM chain (templated per layer shape)
    gemm_conv<LPre0><<<dim3((NPTS / 16) / 8, BATCH), blk, 0, stream>>>(ACT, BWpre0, b_pre0, ACT);
    gemm_conv<LPre1><<<dim3((NPTS / 16) / 8, BATCH), blk, 0, stream>>>(ACT, BWpre1, b_pre1, ACT);
    flags_kernel<<<dim3(NPTS / 256, BATCH), blk, 0, stream>>>(ACT, w_flags, b_flags, out);
    gemm_conv<LMerge><<<dim3((MOUT / 16) / 8, BATCH), blk, 0, stream>>>(ACT, BWmerge, b_merge, S0);
    gemm_conv<LPost><<<dim3((MOUT / 16) / 8, BATCH), blk, 0, stream>>>(S0, BWpost[0], b_post[0], S1);
    gemm_conv<LPost><<<dim3((MOUT / 16) / 8, BATCH), blk, 0, stream>>>(S1, BWpost[1], b_post[1], S0);
    gemm_conv<LPost><<<dim3((MOUT / 16) / 8, BATCH), blk, 0, stream>>>(S0, BWpost[2], b_post[2], S1);
    gemm_conv<LPost><<<dim3((MOUT / 16) / 8, BATCH), blk, 0, stream>>>(S1, BWpost[3], b_post[3], S0);

    // 5. sym head + softmax
    sym_softmax<<<dim3((MOUT / 16) / 8, BATCH), blk, 0, stream>>>(S0, BWsym, b_sym, out);
}
